// TinyViT_44057774523106
// MI455X (gfx1250) — matrix operand order repaired and checked by reading, not compiled, not run on hardware
//
#include <hip/hip_runtime.h>
#include <hip/hip_bf16.h>
#include <math.h>
#include <stdint.h>

// ---------------------------------------------------------------- constants
#define B_ 8
#define L_ 1280
#define D_ 192
#define H_ 6
#define KD_ 32
#define N3_ 576     // 3*D
#define MH_ 768     // MLP hidden
#define M_ (B_ * L_)   // 10240 rows
#define NKC_ (L_ / 32) // 40 key chunks

typedef unsigned short ushort_t;
typedef __attribute__((ext_vector_type(16))) __bf16 v16bf;
typedef __attribute__((ext_vector_type(8)))  float  v8f;
typedef __attribute__((ext_vector_type(8)))  unsigned int v8u;

// ---------------------------------------------------------------- helpers
__device__ __forceinline__ ushort_t f2bf(float f) {
    union { float f; unsigned u; } x; x.f = f;
    unsigned r = x.u + 0x7FFFu + ((x.u >> 16) & 1u);   // round-to-nearest-even
    return (ushort_t)(r >> 16);
}

// A-fragment (16x32 bf16, M x K): lane holds row m = lane%16, h = lane/16.
// element e -> K = kk + 16*(e>=8) + 8*h + (e&7); pairs are K-contiguous.
__device__ __forceinline__ v16bf load_a_frag(const ushort_t* rowptr, int kk, int h) {
    v8u u;
#pragma unroll
    for (int j = 0; j < 8; ++j) {
        int K = kk + ((j >= 4) ? 16 : 0) + 8 * h + 2 * (j & 3);
        u[j] = *(const unsigned int*)(rowptr + K);
    }
    return __builtin_bit_cast(v16bf, u);
}

// B-fragment (32x16 bf16, K x N): lane holds col n = lane%16, h = lane/16,
// element e -> K = kk + 16*(e>=8) + 8*h + (e&7) (same order as A).
__device__ __forceinline__ v16bf load_b_frag(const ushort_t* colptr, int kk, int h) {
    v8u u;
#pragma unroll
    for (int j = 0; j < 8; ++j) {
        int K = kk + ((j >= 4) ? 16 : 0) + 8 * h + 2 * (j & 3);
        u[j] = *(const unsigned int*)(colptr + K);
    }
    return __builtin_bit_cast(v16bf, u);
}

__device__ __forceinline__ v8f wmma_bf16(v16bf a, v16bf b, v8f c) {
    return __builtin_amdgcn_wmma_f32_16x16x32_bf16(false, a, false, b, (short)0, c,
                                                   false, false);
}

// async global->LDS 128-bit copy (per-lane), tracked by ASYNCcnt
__device__ __forceinline__ void async_b128(unsigned lds_off, const void* src) {
    asm volatile("global_load_async_to_lds_b128 %0, %1, off"
                 :: "v"(lds_off), "v"(src) : "memory");
}
__device__ __forceinline__ void wait_async0() {
    asm volatile("s_wait_asynccnt 0x0" ::: "memory");
}

// GEMM core: one wave computes a 16 x (16*NT) tile, A-fragment reused NT times.
// Software-pipelined one K-step ahead: loads for kk+32 are issued before the
// WMMAs of kk so the load clause overlaps matrix math (distinct registers,
// no full s_wait_loadcnt 0 before each WMMA).
template <int NT, int KDIM>
__device__ __forceinline__ void gemm_core(const ushort_t* __restrict__ arow,
                                          const ushort_t* __restrict__ bcol,
                                          int h, v8f acc[NT]) {
    v16bf a = load_a_frag(arow, 0, h);
    v16bf b[NT];
#pragma unroll
    for (int t = 0; t < NT; ++t) b[t] = load_b_frag(bcol + (size_t)t * 16 * KDIM, 0, h);

#pragma unroll
    for (int kk = 0; kk < KDIM; kk += 32) {
        v16bf an;
        v16bf bn[NT];
        if (kk + 32 < KDIM) {
            an = load_a_frag(arow, kk + 32, h);
#pragma unroll
            for (int t = 0; t < NT; ++t)
                bn[t] = load_b_frag(bcol + (size_t)t * 16 * KDIM, kk + 32, h);
        } else {
            an = a;
#pragma unroll
            for (int t = 0; t < NT; ++t) bn[t] = b[t];
        }
#pragma unroll
        for (int t = 0; t < NT; ++t) acc[t] = wmma_bf16(a, b[t], acc[t]);
        a = an;
#pragma unroll
        for (int t = 0; t < NT; ++t) b[t] = bn[t];
    }
}

// ------------------------------------------------- weight convert+transpose
// dst[n*K + k] = bf16(src[k*N + n])   (src is row-major [K][N])
__global__ __launch_bounds__(256) void transpose_bf16_k(const float* __restrict__ src,
                                                        ushort_t* __restrict__ dst,
                                                        int K, int N) {
    int idx = blockIdx.x * 256 + threadIdx.x;
    if (idx >= K * N) return;
    int n = idx / K, k = idx % K;
    dst[idx] = f2bf(src[(size_t)k * N + n]);
}

// ------------------------------------------------------------- layernorm
// one wave32 per 192-element row; lane handles 6 contiguous cols
__global__ __launch_bounds__(256) void ln_k(const float* __restrict__ x,
                                            const float* __restrict__ g,
                                            const float* __restrict__ b,
                                            ushort_t* __restrict__ out) {
    int wave = threadIdx.x >> 5, lane = threadIdx.x & 31;
    int row = blockIdx.x * 8 + wave;
    const float* xr = x + (size_t)row * D_ + lane * 6;
    float v[6];
    float s = 0.f;
#pragma unroll
    for (int i = 0; i < 6; ++i) { v[i] = xr[i]; s += v[i]; }
#pragma unroll
    for (int m = 1; m < 32; m <<= 1) s += __shfl_xor(s, m, 32);
    float mean = s * (1.0f / D_);
    float q = 0.f;
#pragma unroll
    for (int i = 0; i < 6; ++i) { float d = v[i] - mean; q += d * d; }
#pragma unroll
    for (int m = 1; m < 32; m <<= 1) q += __shfl_xor(q, m, 32);
    float rstd = rsqrtf(q * (1.0f / D_) + 1e-5f);
    ushort_t* orow = out + (size_t)row * D_ + lane * 6;
#pragma unroll
    for (int i = 0; i < 6; ++i) {
        int c = lane * 6 + i;
        orow[i] = f2bf((v[i] - mean) * rstd * g[c] + b[c]);
    }
}

// ------------------------------------------------------------- QKV GEMM
// C[M x 576] = h1 @ wqkvT ; scatter into q,k [BH][L][32] and vT [BH][32][L]
__global__ __launch_bounds__(256) void qkv_gemm_k(const ushort_t* __restrict__ h1,
                                                  const ushort_t* __restrict__ wT,
                                                  const float* __restrict__ bias,
                                                  ushort_t* __restrict__ q,
                                                  ushort_t* __restrict__ kout,
                                                  ushort_t* __restrict__ vT) {
    constexpr int NT = 4;
    int wave = threadIdx.x >> 5, lane = threadIdx.x & 31;
    int laneg = lane & 15, h = lane >> 4;
    int tile = blockIdx.x * 8 + wave;
    constexpr int TN = N3_ / (16 * NT);
    int tm = tile / TN, tn = tile % TN;
    int m0 = tm * 16, n0 = tn * 16 * NT + laneg;

    const ushort_t* arow = h1 + (size_t)(m0 + laneg) * D_;
    const ushort_t* bcol = wT + (size_t)n0 * D_;
    v8f acc[NT] = {};
    gemm_core<NT, D_>(arow, bcol, h, acc);

#pragma unroll
    for (int t = 0; t < NT; ++t) {
        int n = n0 + t * 16;
        float bn = bias[n];
        int head = n / 96, r = n % 96, sec = r / 32, d = r % 32;
#pragma unroll
        for (int vr = 0; vr < 8; ++vr) {
            int row = m0 + vr + 8 * h;
            int bb = row / L_, l = row % L_;
            int bh = bb * H_ + head;
            ushort_t val = f2bf(acc[t][vr] + bn);
            if (sec == 0)      q[((size_t)bh * L_ + l) * KD_ + d] = val;
            else if (sec == 1) kout[((size_t)bh * L_ + l) * KD_ + d] = val;
            else               vT[((size_t)bh * KD_ + d) * L_ + l] = val;
        }
    }
}

// ------------------------------------------------------------- attention
// block = (qblock, head, batch); 8 waves x 16 query rows; flash-style.
// K/V chunks staged into LDS with double-buffered async copies.
__global__ __launch_bounds__(256) void attn_k(const ushort_t* __restrict__ q,
                                              const ushort_t* __restrict__ kmat,
                                              const ushort_t* __restrict__ vT,
                                              const int* __restrict__ bias_idx,
                                              const float* __restrict__ bias_table,
                                              int n_off,
                                              ushort_t* __restrict__ o) {
    __shared__ __align__(16) ushort_t kbuf[2][32][32];  // [buf][key][ch]
    __shared__ __align__(16) ushort_t vbuf[2][32][32];  // [buf][ch][key]
    __shared__ __align__(16) ushort_t pl[8][16][32];    // per-wave P tile
    int tid = threadIdx.x;
    int wave = tid >> 5, lane = tid & 31;
    int laneg = lane & 15, h = lane >> 4;
    int head = blockIdx.y, bb = blockIdx.z;
    int bh = bb * H_ + head;
    int qbase = blockIdx.x * 128 + wave * 16;
    const float scale = 0.17677669529663689f;  // 32^-0.5

    const ushort_t* qb = q + (size_t)bh * L_ * KD_;
    const ushort_t* kb = kmat + (size_t)bh * L_ * KD_;
    const ushort_t* vb = vT + (size_t)bh * KD_ * L_;
    const float* btab = bias_table + (size_t)head * n_off;

    // per-thread async-copy slot: threads 0..127 -> K chunk, 128..255 -> V chunk
    auto issue = [&](int kc) {
        int buf = kc & 1;
        int kbase = kc * 32;
        const ushort_t* src;
        unsigned dst;
        if (tid < 128) {
            int r = tid >> 2, seg = tid & 3;
            src = kb + (size_t)(kbase + r) * KD_ + seg * 8;
            dst = (unsigned)(uintptr_t)&kbuf[buf][r][seg * 8];
        } else {
            int t2 = tid - 128;
            int ch = t2 >> 2, seg = t2 & 3;
            src = vb + (size_t)ch * L_ + kbase + seg * 8;
            dst = (unsigned)(uintptr_t)&vbuf[buf][ch][seg * 8];
        }
        async_b128(dst, src);
    };

    v16bf aQ = load_a_frag(qb + (size_t)(qbase + laneg) * KD_, 0, h);
    v8f acc0 = {}, acc1 = {};
    float mrow[8], lrow[8];
#pragma unroll
    for (int i = 0; i < 8; ++i) { mrow[i] = -1e30f; lrow[i] = 0.f; }

    issue(0);
    wait_async0();
    __syncthreads();

    for (int kc = 0; kc < NKC_; ++kc) {
        if (kc + 1 < NKC_) issue(kc + 1);   // prefetch next chunk (other buffer)
        int buf = kc & 1;
        int kbase = kc * 32;

        v16bf bK0 = load_b_frag(&kbuf[buf][laneg][0], 0, h);
        v16bf bK1 = load_b_frag(&kbuf[buf][16 + laneg][0], 0, h);
        v8f z = {};
        v8f s0 = wmma_bf16(aQ, bK0, z);
        v8f s1 = wmma_bf16(aQ, bK1, z);

#pragma unroll
        for (int vr = 0; vr < 8; ++vr) {
            int qrow = qbase + vr + 8 * h;
            const int* ir = bias_idx + (size_t)qrow * L_ + kbase;
            s0[vr] = s0[vr] * scale + btab[ir[laneg]];
            s1[vr] = s1[vr] * scale + btab[ir[16 + laneg]];

            float mx = fmaxf(s0[vr], s1[vr]);
#pragma unroll
            for (int m = 1; m < 16; m <<= 1) mx = fmaxf(mx, __shfl_xor(mx, m, 32));
            float mnew = fmaxf(mrow[vr], mx);
            float corr = __expf(mrow[vr] - mnew);
            float p0 = __expf(s0[vr] - mnew);
            float p1 = __expf(s1[vr] - mnew);
            float rs = p0 + p1;
#pragma unroll
            for (int m = 1; m < 16; m <<= 1) rs += __shfl_xor(rs, m, 32);
            lrow[vr] = lrow[vr] * corr + rs;
            mrow[vr] = mnew;
            acc0[vr] *= corr;
            acc1[vr] *= corr;
            pl[wave][vr + 8 * h][laneg]      = f2bf(p0);
            pl[wave][vr + 8 * h][16 + laneg] = f2bf(p1);
        }
        // pl is wave-private; per-wave DS ops are in-order -> no barrier needed
        v16bf aP = load_a_frag(&pl[wave][laneg][0], 0, h);
        v16bf bV0 = load_b_frag(&vbuf[buf][laneg][0], 0, h);
        v16bf bV1 = load_b_frag(&vbuf[buf][16 + laneg][0], 0, h);
        acc0 = wmma_bf16(aP, bV0, acc0);
        acc1 = wmma_bf16(aP, bV1, acc1);

        wait_async0();       // prefetch of chunk kc+1 has landed
        __syncthreads();     // all waves done with buf before it is reused
    }

#pragma unroll
    for (int vr = 0; vr < 8; ++vr) {
        int qrow = qbase + vr + 8 * h;
        float inv = 1.0f / lrow[vr];
        ushort_t* orow = o + ((size_t)bb * L_ + qrow) * D_ + head * KD_;
        orow[laneg]      = f2bf(acc0[vr] * inv);
        orow[16 + laneg] = f2bf(acc1[vr] * inv);
    }
}

// ------------------------------------------------- proj GEMM + residual (f32)
__global__ __launch_bounds__(256) void proj_gemm_k(const ushort_t* __restrict__ o,
                                                   const ushort_t* __restrict__ wT,
                                                   const float* __restrict__ bias,
                                                   const float* __restrict__ xz,
                                                   float* __restrict__ xm) {
    constexpr int NT = 2;
    int wave = threadIdx.x >> 5, lane = threadIdx.x & 31;
    int laneg = lane & 15, h = lane >> 4;
    int tile = blockIdx.x * 8 + wave;
    constexpr int TN = D_ / (16 * NT);
    int tm = tile / TN, tn = tile % TN;
    int m0 = tm * 16, n0 = tn * 16 * NT + laneg;

    v8f acc[NT] = {};
    gemm_core<NT, D_>(o + (size_t)(m0 + laneg) * D_, wT + (size_t)n0 * D_, h, acc);
#pragma unroll
    for (int t = 0; t < NT; ++t) {
        int n = n0 + t * 16;
        float bn = bias[n];
#pragma unroll
        for (int vr = 0; vr < 8; ++vr) {
            size_t row = m0 + vr + 8 * h;
            xm[row * D_ + n] = xz[row * D_ + n] + acc[t][vr] + bn;
        }
    }
}

// ------------------------------------------- depthwise 3x3 conv + BN affine
__global__ __launch_bounds__(256) void conv_bn_k(const float* __restrict__ xm,
                                                 const float* __restrict__ w,
                                                 const float* __restrict__ sc,
                                                 const float* __restrict__ bi,
                                                 float* __restrict__ xm2) {
    int idx = blockIdx.x * 256 + threadIdx.x;
    int c = idx % D_, n = idx / D_;
    int bb = n / L_, l = n % L_;
    int Hh, Ww, base;
    if (l < 1024) { Hh = 32; Ww = 32; base = 0; }
    else          { Hh = 16; Ww = 16; base = 1024; l -= 1024; }
    int y = l / Ww, x = l % Ww;
    float acc = 0.f;
#pragma unroll
    for (int ky = 0; ky < 3; ++ky) {
#pragma unroll
        for (int kx = 0; kx < 3; ++kx) {
            int yy = y + ky - 1, xx = x + kx - 1;
            if (yy >= 0 && yy < Hh && xx >= 0 && xx < Ww)
                acc += xm[((size_t)bb * L_ + base + yy * Ww + xx) * D_ + c] *
                       w[(ky * 3 + kx) * D_ + c];
        }
    }
    xm2[idx] = acc * sc[c] + bi[c];
}

// -------------------------------------------------- MLP GEMM1 + exact GELU
__global__ __launch_bounds__(256) void mlp1_k(const ushort_t* __restrict__ h2,
                                              const ushort_t* __restrict__ wT,
                                              const float* __restrict__ bias,
                                              ushort_t* __restrict__ hid) {
    constexpr int NT = 4;
    int wave = threadIdx.x >> 5, lane = threadIdx.x & 31;
    int laneg = lane & 15, h = lane >> 4;
    int tile = blockIdx.x * 8 + wave;
    constexpr int TN = MH_ / (16 * NT);
    int tm = tile / TN, tn = tile % TN;
    int m0 = tm * 16, n0 = tn * 16 * NT + laneg;

    v8f acc[NT] = {};
    gemm_core<NT, D_>(h2 + (size_t)(m0 + laneg) * D_, wT + (size_t)n0 * D_, h, acc);
#pragma unroll
    for (int t = 0; t < NT; ++t) {
        int n = n0 + t * 16;
        float bn = bias[n];
#pragma unroll
        for (int vr = 0; vr < 8; ++vr) {
            size_t row = m0 + vr + 8 * h;
            float x = acc[t][vr] + bn;
            float g = 0.5f * x * (1.0f + erff(x * 0.70710678118654752f));
            hid[row * MH_ + n] = f2bf(g);
        }
    }
}

// -------------------------------------------------- MLP GEMM2 + residual
__global__ __launch_bounds__(256) void mlp2_k(const ushort_t* __restrict__ hid,
                                              const ushort_t* __restrict__ wT,
                                              const float* __restrict__ bias,
                                              const float* __restrict__ xm2,
                                              float* __restrict__ out) {
    constexpr int NT = 2;
    int wave = threadIdx.x >> 5, lane = threadIdx.x & 31;
    int laneg = lane & 15, h = lane >> 4;
    int tile = blockIdx.x * 8 + wave;
    constexpr int TN = D_ / (16 * NT);
    int tm = tile / TN, tn = tile % TN;
    int m0 = tm * 16, n0 = tn * 16 * NT + laneg;

    v8f acc[NT] = {};
    gemm_core<NT, MH_>(hid + (size_t)(m0 + laneg) * MH_, wT + (size_t)n0 * MH_, h, acc);
#pragma unroll
    for (int t = 0; t < NT; ++t) {
        int n = n0 + t * 16;
        float bn = bias[n];
#pragma unroll
        for (int vr = 0; vr < 8; ++vr) {
            size_t row = m0 + vr + 8 * h;
            out[row * D_ + n] = xm2[row * D_ + n] + acc[t][vr] + bn;
        }
    }
}

// ---------------------------------------------------------------- launcher
extern "C" void kernel_launch(void* const* d_in, const int* in_sizes, int n_in,
                              void* d_out, int out_size, void* d_ws, size_t ws_size,
                              hipStream_t stream) {
    const float* xz      = (const float*)d_in[0];
    const int*   biasidx = (const int*)d_in[1];
    const float* ln1_g   = (const float*)d_in[2];
    const float* ln1_b   = (const float*)d_in[3];
    const float* wqkv    = (const float*)d_in[4];
    const float* bqkv    = (const float*)d_in[5];
    const float* wproj   = (const float*)d_in[6];
    const float* bproj   = (const float*)d_in[7];
    const float* btab    = (const float*)d_in[8];
    const float* convw   = (const float*)d_in[9];
    const float* bns     = (const float*)d_in[10];
    const float* bnb     = (const float*)d_in[11];
    const float* ln2_g   = (const float*)d_in[12];
    const float* ln2_b   = (const float*)d_in[13];
    const float* w1      = (const float*)d_in[14];
    const float* b1      = (const float*)d_in[15];
    const float* w2      = (const float*)d_in[16];
    const float* b2      = (const float*)d_in[17];
    int n_off = in_sizes[8] / H_;

    char* ws = (char*)d_ws;
    size_t off = 0;
    auto alloc = [&](size_t bytes) {
        char* p = ws + off;
        off += (bytes + 255) & ~(size_t)255;
        return p;
    };
    ushort_t* wqkvT  = (ushort_t*)alloc((size_t)N3_ * D_ * 2);
    ushort_t* wprojT = (ushort_t*)alloc((size_t)D_ * D_ * 2);
    ushort_t* w1T    = (ushort_t*)alloc((size_t)MH_ * D_ * 2);
    ushort_t* w2T    = (ushort_t*)alloc((size_t)D_ * MH_ * 2);
    ushort_t* h1     = (ushort_t*)alloc((size_t)M_ * D_ * 2);
    ushort_t* qd     = (ushort_t*)alloc((size_t)B_ * H_ * L_ * KD_ * 2);
    ushort_t* kd     = (ushort_t*)alloc((size_t)B_ * H_ * L_ * KD_ * 2);
    ushort_t* vTd    = (ushort_t*)alloc((size_t)B_ * H_ * KD_ * L_ * 2);
    ushort_t* od     = (ushort_t*)alloc((size_t)M_ * D_ * 2);
    float*    xm     = (float*)alloc((size_t)M_ * D_ * 4);
    float*    xm2    = (float*)alloc((size_t)M_ * D_ * 4);
    ushort_t* h2     = (ushort_t*)alloc((size_t)M_ * D_ * 2);
    ushort_t* hid    = (ushort_t*)alloc((size_t)M_ * MH_ * 2);

    // weight convert + transpose
    transpose_bf16_k<<<(D_ * N3_ + 255) / 256, 256, 0, stream>>>(wqkv, wqkvT, D_, N3_);
    transpose_bf16_k<<<(D_ * D_ + 255) / 256, 256, 0, stream>>>(wproj, wprojT, D_, D_);
    transpose_bf16_k<<<(D_ * MH_ + 255) / 256, 256, 0, stream>>>(w1, w1T, D_, MH_);
    transpose_bf16_k<<<(MH_ * D_ + 255) / 256, 256, 0, stream>>>(w2, w2T, MH_, D_);

    // LN1 -> bf16
    ln_k<<<M_ / 8, 256, 0, stream>>>(xz, ln1_g, ln1_b, h1);
    // QKV GEMM (WMMA, 16x64 wave tiles, pipelined)
    qkv_gemm_k<<<(M_ / 16) * (N3_ / 64) / 8, 256, 0, stream>>>(h1, wqkvT, bqkv,
                                                               qd, kd, vTd);
    // attention (WMMA + async-LDS double-buffered K/V + online softmax)
    attn_k<<<dim3(L_ / 128, H_, B_), 256, 0, stream>>>(qd, kd, vTd, biasidx, btab,
                                                       n_off, od);
    // proj + residual (16x32 wave tiles)
    proj_gemm_k<<<(M_ / 16) * (D_ / 32) / 8, 256, 0, stream>>>(od, wprojT, bproj,
                                                               xz, xm);
    // depthwise conv + BN
    conv_bn_k<<<(M_ * D_) / 256, 256, 0, stream>>>(xm, convw, bns, bnb, xm2);
    // LN2 -> bf16
    ln_k<<<M_ / 8, 256, 0, stream>>>(xm2, ln2_g, ln2_b, h2);
    // MLP
    mlp1_k<<<(M_ / 16) * (MH_ / 64) / 8, 256, 0, stream>>>(h2, w1T, b1, hid);
    mlp2_k<<<(M_ / 16) * (D_ / 32) / 8, 256, 0, stream>>>(hid, w2T, b2, xm2,
                                                          (float*)d_out);
}
